// TransformerBlock_23691039605035
// MI455X (gfx1250) — matrix operand order repaired and checked by reading, not compiled, not run on hardware
//
#include <hip/hip_runtime.h>

// ---------------------------------------------------------------------------
// TransformerBlock forward for MI455X (gfx1250), wave32 + WMMA bf16.
// B=8, T=2048, D=64, H=4, DH=16, FF=256.
// ---------------------------------------------------------------------------

#define BQ 8
#define TQ 2048
#define DQ 64
#define HQ 4
#define DHQ 16
#define FFQ 256
#define BT (BQ * TQ)          // 16384 rows

typedef __attribute__((ext_vector_type(16))) __bf16        v16bf;
typedef __attribute__((ext_vector_type(8)))  float         v8f;
typedef __attribute__((ext_vector_type(8)))  unsigned short v8u;
typedef __attribute__((ext_vector_type(16))) unsigned short v16u;

static __device__ __forceinline__ unsigned short f2bf(float f) {
    unsigned u = __builtin_bit_cast(unsigned, f);
    u += 0x7FFFu + ((u >> 16) & 1u);      // round-to-nearest-even
    return (unsigned short)(u >> 16);
}

static __device__ __forceinline__ v8f zero8() {
    v8f z = {0.f, 0.f, 0.f, 0.f, 0.f, 0.f, 0.f, 0.f};
    return z;
}

// A-operand (16x32 bf16, row-major source): lane row = lane%16,
// element e -> K = k0 + 16g + e
static __device__ __forceinline__ v16bf load_a32(const unsigned short* rowPtr, int k0) {
    int g = (threadIdx.x & 31) >> 4;
    v8u lo = *(const v8u*)(rowPtr + k0 + 16 * g);
    v8u hi = *(const v8u*)(rowPtr + k0 + 16 * g + 8);
    v16u t;
#pragma unroll
    for (int i = 0; i < 8; ++i) { t[i] = lo[i]; t[i + 8] = hi[i]; }
    return __builtin_bit_cast(v16bf, t);
}

// A-operand with real K=16, padded to 32 with zeros (K>=16 are zero).
static __device__ __forceinline__ v16bf load_a16pad(const unsigned short* rowPtr) {
    int g = (threadIdx.x & 31) >> 4;
    v8u lo = *(const v8u*)(rowPtr);
    v8u hi = *(const v8u*)(rowPtr + 8);
    v16u t;
#pragma unroll
    for (int i = 0; i < 8; ++i) { t[i] = (g == 0) ? lo[i] : (unsigned short)0; t[i + 8] = (g == 0) ? hi[i] : (unsigned short)0; }
    return __builtin_bit_cast(v16bf, t);
}

// B-operand (32x16 bf16, stored col-major i.e. [n][k]): lane col = lane%16,
// element e -> K = k0 + 16g + e  => one contiguous 32-byte load.
static __device__ __forceinline__ v16bf load_b32(const unsigned short* rowPtr, int k0) {
    int g = (threadIdx.x & 31) >> 4;
    return __builtin_bit_cast(v16bf, *(const v16u*)(rowPtr + k0 + 16 * g));
}

// B-operand with real K=16 padded to 32: lanes 16..31 are all pad (zero).
static __device__ __forceinline__ v16bf load_b16pad(const unsigned short* rowPtr) {
    int g = (threadIdx.x & 31) >> 4;
    v16u t;
    if (g == 0) {
        t = *(const v16u*)(rowPtr);
    } else {
#pragma unroll
        for (int i = 0; i < 16; ++i) t[i] = 0;
    }
    return __builtin_bit_cast(v16bf, t);
}

static __device__ __forceinline__ v8f wmma_bf16(v16bf a, v16bf b, v8f c) {
    return __builtin_amdgcn_wmma_f32_16x16x32_bf16(false, a, false, b, (short)0, c,
                                                   false, false);
}

// ---------------------------------------------------------------------------
// Weight pack: WT[b][n][k] = bf16(W[b][k][n])
// ---------------------------------------------------------------------------
__global__ void pack_wT_k(const float* __restrict__ W, unsigned short* __restrict__ WT,
                          int K, int N, int total) {
    int idx = blockIdx.x * blockDim.x + threadIdx.x;
    if (idx >= total) return;
    int kn = K * N;
    int bidx = idx / kn;
    int rem  = idx - bidx * kn;
    int n  = rem / K;
    int kk = rem - n * K;
    WT[(size_t)bidx * kn + (size_t)n * K + kk] =
        f2bf(W[(size_t)bidx * kn + (size_t)kk * N + n]);
}

// ---------------------------------------------------------------------------
// LayerNorm over D=64; 1 wave per row, 8 rows per 256-thread block.
// ---------------------------------------------------------------------------
__global__ __launch_bounds__(256) void ln_k(const float* __restrict__ x,
                                            const float* __restrict__ w,
                                            const float* __restrict__ bias,
                                            unsigned short* __restrict__ out) {
    int lane = threadIdx.x & 31;
    int row  = blockIdx.x * 8 + (threadIdx.x >> 5);
    const float* xr = x + (size_t)row * DQ;
    float2 v = ((const float2*)xr)[lane];
    float s = v.x + v.y;
#pragma unroll
    for (int d = 16; d >= 1; d >>= 1) s += __shfl_xor(s, d, 32);
    float mu = s * (1.f / 64.f);
    float d0 = v.x - mu, d1 = v.y - mu;
    float q = d0 * d0 + d1 * d1;
#pragma unroll
    for (int d = 16; d >= 1; d >>= 1) q += __shfl_xor(q, d, 32);
    float rstd = rsqrtf(q * (1.f / 64.f) + 1e-5f);
    float y0 = d0 * rstd * w[2 * lane]     + bias[2 * lane];
    float y1 = d1 * rstd * w[2 * lane + 1] + bias[2 * lane + 1];
    ((unsigned*)(out + (size_t)row * DQ))[lane] =
        (unsigned)f2bf(y0) | ((unsigned)f2bf(y1) << 16);
}

// ---------------------------------------------------------------------------
// QKV projection: per wave one 16(t) x 16(dh) tile for q, k, v of one head.
// q,k stored [bh][t][dh]; v stored transposed [bh][dh][t].
// ---------------------------------------------------------------------------
__global__ __launch_bounds__(128) void qkv_k(const unsigned short* __restrict__ hbf,
                                             const unsigned short* __restrict__ wqT,
                                             const unsigned short* __restrict__ wkT,
                                             const unsigned short* __restrict__ wvT,
                                             unsigned short* __restrict__ q,
                                             unsigned short* __restrict__ k,
                                             unsigned short* __restrict__ vT) {
    int tile = blockIdx.x * 4 + (threadIdx.x >> 5);  // bh*128 + tb
    int tb = tile & 127;
    int bh = tile >> 7;               // b*H + h
    int h  = bh & 3;
    int b  = bh >> 2;
    int lane = threadIdx.x & 31, ln16 = lane & 15, g = lane >> 4;

    const unsigned short* hrow = hbf + ((size_t)(b * TQ + tb * 16 + ln16)) * DQ;
    v16bf a0 = load_a32(hrow, 0);
    v16bf a1 = load_a32(hrow, 32);

    // ---- Q ----
    const unsigned short* wr = wqT + ((size_t)h * DHQ + ln16) * DQ;
    v8f cq = zero8();
    cq = wmma_bf16(a0, load_b32(wr, 0), cq);
    cq = wmma_bf16(a1, load_b32(wr, 32), cq);
    unsigned short* qb = q + ((size_t)bh * TQ + tb * 16) * DHQ + ln16;
#pragma unroll
    for (int r = 0; r < 8; ++r) qb[(r + 8 * g) * DHQ] = f2bf(cq[r]);

    // ---- K ----
    wr = wkT + ((size_t)h * DHQ + ln16) * DQ;
    v8f ck = zero8();
    ck = wmma_bf16(a0, load_b32(wr, 0), ck);
    ck = wmma_bf16(a1, load_b32(wr, 32), ck);
    unsigned short* kb = k + ((size_t)bh * TQ + tb * 16) * DHQ + ln16;
#pragma unroll
    for (int r = 0; r < 8; ++r) kb[(r + 8 * g) * DHQ] = f2bf(ck[r]);

    // ---- V (store transposed: row dh = ln16, t contiguous) ----
    wr = wvT + ((size_t)h * DHQ + ln16) * DQ;
    v8f cv = zero8();
    cv = wmma_bf16(a0, load_b32(wr, 0), cv);
    cv = wmma_bf16(a1, load_b32(wr, 32), cv);
    v8u pv;
#pragma unroll
    for (int r = 0; r < 8; ++r) pv[r] = f2bf(cv[r]);
    *(v8u*)(vT + ((size_t)bh * DHQ + ln16) * TQ + tb * 16 + 8 * g) = pv;
}

// ---------------------------------------------------------------------------
// Flash attention per (b,h, 16-query block). Computes S^T = K*Q^T so queries
// sit in C-layout columns; O^T = V^T * P^T accumulated in registers.
// ---------------------------------------------------------------------------
__global__ __launch_bounds__(128) void attn_k(const unsigned short* __restrict__ q,
                                              const unsigned short* __restrict__ k,
                                              const unsigned short* __restrict__ vT,
                                              unsigned short* __restrict__ obuf) {
    int tile = blockIdx.x * 4 + (threadIdx.x >> 5);
    int qb = tile & 127;
    int bh = tile >> 7;
    int b = bh >> 2, h = bh & 3;
    int lane = threadIdx.x & 31, ln16 = lane & 15, g = lane >> 4;
    int t_q = qb * 16 + ln16;

    v16bf bq = load_b16pad(q + ((size_t)bh * TQ + t_q) * DHQ);   // Q^T as B operand
    const unsigned short* kbh = k + (size_t)bh * TQ * DHQ;
    const unsigned short* vbh = vT + ((size_t)bh * DHQ + ln16) * TQ;

    float m = -3.0e38f, l = 0.f;
    v8f o = zero8();

    for (int kbk = 0; kbk <= qb; ++kbk) {
        v16bf ak = load_a16pad(kbh + (size_t)(kbk * 16 + ln16) * DHQ);
        v8f s = wmma_bf16(ak, bq, zero8());    // S^T tile: row=key(r+8g), col=query

        float pv[8];
#pragma unroll
        for (int r = 0; r < 8; ++r) {
            float sv = s[r] * 0.25f;           // 1/sqrt(16)
            if (kbk == qb && (r + 8 * g) > ln16) sv = -3.0e38f;  // causal
            pv[r] = sv;
        }
        float lm = pv[0];
#pragma unroll
        for (int r = 1; r < 8; ++r) lm = fmaxf(lm, pv[r]);
        lm = fmaxf(lm, __shfl_xor(lm, 16, 32));
        float nm    = fmaxf(m, lm);
        float alpha = __expf(m - nm);
        float ls    = 0.f;
#pragma unroll
        for (int r = 0; r < 8; ++r) { pv[r] = __expf(pv[r] - nm); ls += pv[r]; }
        ls += __shfl_xor(ls, 16, 32);
        l = l * alpha + ls;
        m = nm;
#pragma unroll
        for (int r = 0; r < 8; ++r) o[r] *= alpha;

        // Build P^T as B operand: g==0 lanes hold keys 0..15 of their query
        // (own half = keys 0..7, partner half = keys 8..15); g==1 lanes = pad.
        v16u pb;
#pragma unroll
        for (int r = 0; r < 8; ++r) {
            float pp = __shfl_xor(pv[r], 16, 32);
            pb[r]     = (g == 0) ? f2bf(pv[r]) : (unsigned short)0;
            pb[r + 8] = (g == 0) ? f2bf(pp)    : (unsigned short)0;
        }
        v16bf av = load_a16pad(vbh + kbk * 16);   // V^T: row dh, keys in K-dim
        o = wmma_bf16(av, __builtin_bit_cast(v16bf, pb), o);
    }

    float invl = 1.0f / l;
    v8u ov;
#pragma unroll
    for (int r = 0; r < 8; ++r) ov[r] = f2bf(o[r] * invl);
    // O^T element r -> dh = 8g + r (contiguous), col = t_q
    *(v8u*)(obuf + ((size_t)(b * TQ + t_q)) * DQ + h * DHQ + 8 * g) = ov;
}

// ---------------------------------------------------------------------------
// Output projection + residual: x2 = x + obuf @ Wp + bp
// ---------------------------------------------------------------------------
__global__ __launch_bounds__(128) void proj_k(const unsigned short* __restrict__ obuf,
                                              const unsigned short* __restrict__ wpT,
                                              const float* __restrict__ bp,
                                              const float* __restrict__ x,
                                              float* __restrict__ x2) {
    int tile = blockIdx.x * 4 + (threadIdx.x >> 5);
    int nb = tile & 3, mb = tile >> 2;
    int lane = threadIdx.x & 31, ln16 = lane & 15, g = lane >> 4;
    const unsigned short* arow = obuf + ((size_t)(mb * 16 + ln16)) * DQ;
    const unsigned short* brow = wpT + ((size_t)(nb * 16 + ln16)) * DQ;
    v8f c = zero8();
    c = wmma_bf16(load_a32(arow, 0),  load_b32(brow, 0),  c);
    c = wmma_bf16(load_a32(arow, 32), load_b32(brow, 32), c);
    int n = nb * 16 + ln16;
    float bias = bp[n];
#pragma unroll
    for (int r = 0; r < 8; ++r) {
        size_t idx = (size_t)(mb * 16 + r + 8 * g) * DQ + n;
        x2[idx] = x[idx] + c[r] + bias;
    }
}

// ---------------------------------------------------------------------------
// FFN layer 1: ff = relu(h2 @ W1 + b1), bf16 out [BT, 256]
// ---------------------------------------------------------------------------
__global__ __launch_bounds__(128) void ffn1_k(const unsigned short* __restrict__ h2,
                                              const unsigned short* __restrict__ w1T,
                                              const float* __restrict__ b1,
                                              unsigned short* __restrict__ ff) {
    int tile = blockIdx.x * 4 + (threadIdx.x >> 5);
    int nb = tile & 15, mb = tile >> 4;
    int lane = threadIdx.x & 31, ln16 = lane & 15, g = lane >> 4;
    const unsigned short* arow = h2 + ((size_t)(mb * 16 + ln16)) * DQ;
    const unsigned short* brow = w1T + ((size_t)(nb * 16 + ln16)) * DQ;
    v8f c = zero8();
    c = wmma_bf16(load_a32(arow, 0),  load_b32(brow, 0),  c);
    c = wmma_bf16(load_a32(arow, 32), load_b32(brow, 32), c);
    int n = nb * 16 + ln16;
    float bias = b1[n];
#pragma unroll
    for (int r = 0; r < 8; ++r) {
        float y = fmaxf(c[r] + bias, 0.f);
        ff[(size_t)(mb * 16 + r + 8 * g) * FFQ + n] = f2bf(y);
    }
}

// ---------------------------------------------------------------------------
// FFN layer 2 + residual: out = x2 + ff @ W2 + b2  (final output, f32)
// ---------------------------------------------------------------------------
__global__ __launch_bounds__(128) void ffn2_k(const unsigned short* __restrict__ ff,
                                              const unsigned short* __restrict__ w2T,
                                              const float* __restrict__ b2,
                                              const float* __restrict__ x2,
                                              float* __restrict__ out) {
    int tile = blockIdx.x * 4 + (threadIdx.x >> 5);
    int nb = tile & 3, mb = tile >> 2;
    int lane = threadIdx.x & 31, ln16 = lane & 15, g = lane >> 4;
    const unsigned short* arow = ff + ((size_t)(mb * 16 + ln16)) * FFQ;
    const unsigned short* brow = w2T + ((size_t)(nb * 16 + ln16)) * FFQ;
    v8f c = zero8();
#pragma unroll
    for (int k0 = 0; k0 < FFQ; k0 += 32)
        c = wmma_bf16(load_a32(arow, k0), load_b32(brow, k0), c);
    int n = nb * 16 + ln16;
    float bias = b2[n];
#pragma unroll
    for (int r = 0; r < 8; ++r) {
        size_t idx = (size_t)(mb * 16 + r + 8 * g) * DQ + n;
        out[idx] = x2[idx] + c[r] + bias;
    }
}

// ---------------------------------------------------------------------------
// Host launcher
// ---------------------------------------------------------------------------
extern "C" void kernel_launch(void* const* d_in, const int* in_sizes, int n_in,
                              void* d_out, int out_size, void* d_ws, size_t ws_size,
                              hipStream_t stream) {
    (void)in_sizes; (void)n_in; (void)out_size; (void)ws_size;

    const float* x     = (const float*)d_in[0];
    const float* ln1_w = (const float*)d_in[1];
    const float* ln1_b = (const float*)d_in[2];
    const float* ln2_w = (const float*)d_in[3];
    const float* ln2_b = (const float*)d_in[4];
    const float* Wq    = (const float*)d_in[5];
    const float* Wk    = (const float*)d_in[6];
    const float* Wv    = (const float*)d_in[7];
    const float* Wp    = (const float*)d_in[8];
    const float* bp    = (const float*)d_in[9];
    const float* W1    = (const float*)d_in[10];
    const float* b1    = (const float*)d_in[11];
    const float* W2    = (const float*)d_in[12];
    const float* b2    = (const float*)d_in[13];
    float* outp        = (float*)d_out;

    char* ws = (char*)d_ws;
    size_t off = 0;
    auto alloc = [&](size_t bytes) -> void* {
        void* p = ws + off;
        off += (bytes + 255) & ~(size_t)255;
        return p;
    };

    unsigned short* hbf  = (unsigned short*)alloc((size_t)BT * DQ * 2);     // LN1 out (bf16)
    unsigned short* qd   = (unsigned short*)alloc((size_t)BQ*HQ*TQ*DHQ*2);  // q [bh][t][dh]
    unsigned short* kd   = (unsigned short*)alloc((size_t)BQ*HQ*TQ*DHQ*2);  // k [bh][t][dh]
    unsigned short* vTd  = (unsigned short*)alloc((size_t)BQ*HQ*DHQ*TQ*2);  // v^T [bh][dh][t]
    unsigned short* obuf = (unsigned short*)alloc((size_t)BT * DQ * 2);     // attn heads concat
    float*          x2   = (float*)alloc((size_t)BT * DQ * 4);              // residual 1
    unsigned short* h2   = (unsigned short*)alloc((size_t)BT * DQ * 2);     // LN2 out
    unsigned short* ffd  = (unsigned short*)alloc((size_t)BT * FFQ * 2);    // relu(ffn1)
    unsigned short* wqT  = (unsigned short*)alloc((size_t)HQ * DQ * DHQ * 2);
    unsigned short* wkT  = (unsigned short*)alloc((size_t)HQ * DQ * DHQ * 2);
    unsigned short* wvT  = (unsigned short*)alloc((size_t)HQ * DQ * DHQ * 2);
    unsigned short* wpT  = (unsigned short*)alloc((size_t)DQ * DQ * 2);
    unsigned short* w1T  = (unsigned short*)alloc((size_t)DQ * FFQ * 2);
    unsigned short* w2T  = (unsigned short*)alloc((size_t)FFQ * DQ * 2);

    // ---- pack weights to transposed bf16 ----
    pack_wT_k<<<(HQ*DQ*DHQ + 255)/256, 256, 0, stream>>>(Wq, wqT, DQ, DHQ, HQ*DQ*DHQ);
    pack_wT_k<<<(HQ*DQ*DHQ + 255)/256, 256, 0, stream>>>(Wk, wkT, DQ, DHQ, HQ*DQ*DHQ);
    pack_wT_k<<<(HQ*DQ*DHQ + 255)/256, 256, 0, stream>>>(Wv, wvT, DQ, DHQ, HQ*DQ*DHQ);
    pack_wT_k<<<(DQ*DQ   + 255)/256, 256, 0, stream>>>(Wp, wpT, DQ, DQ,  DQ*DQ);
    pack_wT_k<<<(DQ*FFQ  + 255)/256, 256, 0, stream>>>(W1, w1T, DQ, FFQ, DQ*FFQ);
    pack_wT_k<<<(FFQ*DQ  + 255)/256, 256, 0, stream>>>(W2, w2T, FFQ, DQ, FFQ*DQ);

    // ---- LN1 ----
    ln_k<<<BT / 8, 256, 0, stream>>>(x, ln1_w, ln1_b, hbf);

    // ---- QKV projections: B*H*(T/16) = 4096 tiles, 4 waves/block ----
    qkv_k<<<1024, 128, 0, stream>>>(hbf, wqT, wkT, wvT, qd, kd, vTd);

    // ---- flash attention: 4096 tiles ----
    attn_k<<<1024, 128, 0, stream>>>(qd, kd, vTd, obuf);

    // ---- output projection + residual: (BT/16)*4 = 4096 tiles ----
    proj_k<<<1024, 128, 0, stream>>>(obuf, wpT, bp, x, x2);

    // ---- LN2 ----
    ln_k<<<BT / 8, 256, 0, stream>>>(x2, ln2_w, ln2_b, h2);

    // ---- FFN1: (BT/16)*16 = 16384 tiles ----
    ffn1_k<<<4096, 128, 0, stream>>>(h2, w1T, b1, ffd);

    // ---- FFN2 + residual: 4096 tiles ----
    ffn2_k<<<1024, 128, 0, stream>>>(ffd, w2T, b2, x2, outp);
}
